// CausalSelfAttention_50646254355207
// MI455X (gfx1250) — matrix operand order repaired and checked by reading, not compiled, not run on hardware
//
#include <hip/hip_runtime.h>
#include <hip/hip_bf16.h>

typedef __bf16 bfloat;
typedef bfloat v16bf __attribute__((ext_vector_type(16)));
typedef bfloat v8bf  __attribute__((ext_vector_type(8)));
typedef float  v8f   __attribute__((ext_vector_type(8)));

#define T_LEN 4096
#define DIMV  1024
#define NHEAD 8
#define HDIM  128

__device__ __forceinline__ bfloat f2bf(float f) {
    __hip_bfloat16 h = __float2bfloat16(f);
    return __builtin_bit_cast(bfloat, h);
}

// ---- WMMA fragment loaders (16x16x32 bf16; contiguous k = 16h + i in both A and B) ------------------
// A-matrix 16x32 (MxK): lane holds row m = row0 + (lane&15); elements i:
//   K = k0 + 16*(lane>>4) + i
__device__ __forceinline__ v16bf load_a(const bfloat* base, int ld, int row0,
                                        int k0, int lane) {
    const int m = row0 + (lane & 15);
    const int b = (lane >> 4) & 1;
    const bfloat* p = base + (size_t)m * ld + k0 + b * 16;
    v8bf lo = *(const v8bf*)p;
    v8bf hi = *(const v8bf*)(p + 8);
    v16bf r;
#pragma unroll
    for (int i = 0; i < 8; i++) { r[i] = lo[i]; r[i + 8] = hi[i]; }
    return r;
}

// B-matrix 32x16 (KxN) loaded from row-major B^T (N x K):
//   lane holds column n = n0 + (lane&15) of B, 16 consecutive K values
//   starting at k0 + 16*(lane>>4): one contiguous 32-byte load.
__device__ __forceinline__ v16bf load_b(const bfloat* baseT, int ld, int n0,
                                        int k0, int lane) {
    const int n = n0 + (lane & 15);
    const int kb = k0 + ((lane >> 4) & 1) * 16;
    return *(const v16bf*)(baseT + (size_t)n * ld + kb);
}

__device__ __forceinline__ v8f wmma_bf16(v16bf a, v16bf b, v8f c) {
    return __builtin_amdgcn_wmma_f32_16x16x32_bf16(
        false, a, false, b, (short)0, c, false, false);
}

// ---- f32 -> bf16 convert --------------------------------------------------
__global__ void cvt_kernel(const float* __restrict__ in, bfloat* __restrict__ out,
                           int n) {
    int i = blockIdx.x * blockDim.x + threadIdx.x;
    if (i < n) out[i] = f2bf(in[i]);
}

// ---- GEMM: C[MxN] f32 = A[MxK] bf16 (row-major) x B[NxK]^T bf16 -----------
// 128-thread block = 4 waves, block tile 128x128, each wave 64x64
// (4 A-frags x 4 B-frags -> 16 WMMAs per 32-deep k-step, ~16 FLOP/B).
// __launch_bounds__(128,1): the wave carries ~200 VGPRs (128 accumulator);
// without this the compiler's occupancy-driven cap spills the C tiles.
__global__ void __launch_bounds__(128, 1)
gemm_nt(const bfloat* __restrict__ A, const bfloat* __restrict__ B,
        float* __restrict__ C, int N, int K) {
    const int lane = threadIdx.x & 31;
    const int w    = threadIdx.x >> 5;          // 0..3
    const int m0 = blockIdx.x * 128 + (w >> 1) * 64;
    const int n0 = blockIdx.y * 128 + (w & 1) * 64;
    v8f acc[4][4] = {};
    for (int k0 = 0; k0 < K; k0 += 32) {
        __builtin_prefetch(A + (size_t)(m0 + (lane & 15)) * K + k0 + 128, 0, 1);
        __builtin_prefetch(B + (size_t)(n0 + (lane & 15)) * K + k0 + 128, 0, 1);
        v16bf a[4];
#pragma unroll
        for (int i = 0; i < 4; i++) a[i] = load_a(A, K, m0 + 16 * i, k0, lane);
#pragma unroll
        for (int j = 0; j < 4; j++) {
            v16bf b = load_b(B, K, n0 + 16 * j, k0, lane);
#pragma unroll
            for (int i = 0; i < 4; i++)
                acc[i][j] = wmma_bf16(a[i], b, acc[i][j]);
        }
    }
    const int n = lane & 15, hb = lane >> 4;
#pragma unroll
    for (int i = 0; i < 4; i++)
#pragma unroll
        for (int j = 0; j < 4; j++)
#pragma unroll
            for (int r = 0; r < 8; r++)
                C[(size_t)(m0 + i * 16 + hb * 8 + r) * N + n0 + 16 * j + n] =
                    acc[i][j][r];
}

// ---- RMS-norm + partial RoPE + V mix; one wave per (t, head-of-24) --------
__global__ void normrope(const float* __restrict__ qkv, const float* __restrict__ ve,
                         const float* __restrict__ lambdas,
                         bfloat* __restrict__ qn, bfloat* __restrict__ kn,
                         bfloat* __restrict__ vt) {
    const int t = blockIdx.x;
    const int hh = blockIdx.y;   // 0..7=q heads, 8..15=k heads, 16..23=v heads
    const int lane = threadIdx.x;
    const float* src = qkv + (size_t)t * (3 * DIMV) + hh * HDIM + lane * 4;
    float xv[4] = {src[0], src[1], src[2], src[3]};
    float ss = xv[0]*xv[0] + xv[1]*xv[1] + xv[2]*xv[2] + xv[3]*xv[3];
#pragma unroll
    for (int off = 1; off < 32; off <<= 1) ss += __shfl_xor(ss, off);
    const float inv = rsqrtf(ss * (1.0f / HDIM) + 1e-6f);
#pragma unroll
    for (int i = 0; i < 4; i++) xv[i] *= inv;

    if (hh < 16) {
        bfloat out[4];
#pragma unroll
        for (int i = 0; i < 4; i++) {
            float part = __shfl_xor(xv[i], 16);  // pairs d <-> d+-64
            int d = lane * 4 + i;
            int j = d & 63;
            float c = 1.0f, s = 0.0f;
            if (j < 32) {
                // inv_freq = (1/1024)^(j/31) = exp(-ln2*10*j/31)
                float invf = __expf((float)j * (-10.0f * 0.69314718056f / 31.0f));
                __sincosf((float)t * invf, &s, &c);
            }
            float yv = (d < 64) ? (xv[i] * c + part * s)   // y1 = x1 c + x2 s
                                : (xv[i] * c - part * s);  // y2 = x2 c - x1 s
            out[i] = f2bf(yv);
        }
        bfloat* dst = (hh < 8 ? qn : kn) + (size_t)t * DIMV + (hh & 7) * HDIM + lane * 4;
        dst[0] = out[0]; dst[1] = out[1]; dst[2] = out[2]; dst[3] = out[3];
    } else {
        const int hv = hh - 16;
        const float l0 = lambdas[0], l1 = lambdas[1];
#pragma unroll
        for (int i = 0; i < 4; i++) {
            int d = lane * 4 + i;
            float vvv = l0 * xv[i] + l1 * ve[(size_t)t * DIMV + hv * HDIM + d];
            vt[(size_t)(hv * HDIM + d) * T_LEN + t] = f2bf(vvv);  // transposed [h][d][t]
        }
    }
}

// ---- Sliding-window attention, one wave per (head, 16-row query tile) -----
__global__ void __launch_bounds__(32, 1)
attn(const bfloat* __restrict__ qn, const bfloat* __restrict__ kn,
     const bfloat* __restrict__ vt, bfloat* __restrict__ y,
     const int* __restrict__ wptr) {
    __shared__ alignas(32) bfloat pls[16 * 32];  // P tile, 16 rows x 32 keys
    const int lane = threadIdx.x;
    const int h = blockIdx.y;
    const int q0 = blockIdx.x * 16;
    const int window = *wptr;
    const bfloat* qh = qn + h * HDIM;
    const bfloat* kh = kn + h * HDIM;
    const bfloat* vh = vt + (size_t)h * HDIM * T_LEN;

    v16bf qf[4];
#pragma unroll
    for (int j = 0; j < 4; j++) qf[j] = load_a(qh, DIMV, q0, j * 32, lane);

    v8f acc[8] = {};
    float mi[8], li[8];
#pragma unroll
    for (int r = 0; r < 8; r++) { mi[r] = -3.0e38f; li[r] = 0.0f; }

    int slo = q0 - (window - 1); if (slo < 0) slo = 0;
    const int n = lane & 15;
    const int hb = lane >> 4;

    for (int sb = slo & ~31; sb < q0 + 16; sb += 32) {
        // scores for 32 keys: two 16x16 tiles, contraction over d=128
        v8f s0 = {}, s1 = {};
#pragma unroll
        for (int kk = 0; kk < 4; kk++) {
            v16bf b0 = load_b(kh, DIMV, sb, kk * 32, lane);
            v16bf b1 = load_b(kh, DIMV, sb + 16, kk * 32, lane);
            s0 = wmma_bf16(qf[kk], b0, s0);
            s1 = wmma_bf16(qf[kk], b1, s1);
        }
        float p0[8], p1[8], alpha[8];
#pragma unroll
        for (int r = 0; r < 8; r++) {
            int m = q0 + r + 8 * hb;
            int d0 = m - (sb + n), d1 = m - (sb + 16 + n);
            float v0 = (d0 >= 0 && d0 < window) ? s0[r] * 0.12f : -3.0e38f;
            float v1 = (d1 >= 0 && d1 < window) ? s1[r] * 0.12f : -3.0e38f;
            float mx = fmaxf(v0, v1);
#pragma unroll
            for (int off = 1; off < 16; off <<= 1) mx = fmaxf(mx, __shfl_xor(mx, off));
            float mnew = fmaxf(mi[r], mx);
            alpha[r] = __expf(mi[r] - mnew);
            float e0 = __expf(v0 - mnew);
            float e1 = __expf(v1 - mnew);
            float rs = e0 + e1;
#pragma unroll
            for (int off = 1; off < 16; off <<= 1) rs += __shfl_xor(rs, off);
            li[r] = li[r] * alpha[r] + rs;
            mi[r] = mnew;
            p0[r] = e0; p1[r] = e1;
        }
#pragma unroll
        for (int dt = 0; dt < 8; dt++)
#pragma unroll
            for (int r = 0; r < 8; r++) acc[dt][r] *= alpha[r];
        // redistribute P (D-layout) -> A-layout via LDS
#pragma unroll
        for (int r = 0; r < 8; r++) {
            int mrow = r + 8 * hb;
            pls[mrow * 32 + n] = f2bf(p0[r]);
            pls[mrow * 32 + 16 + n] = f2bf(p1[r]);
        }
        __syncthreads();
        v16bf pf = load_a(pls, 32, 0, 0, lane);
#pragma unroll
        for (int dt = 0; dt < 8; dt++) {
            v16bf vb = load_b(vh, T_LEN, dt * 16, sb, lane);  // V^T rows = d
            acc[dt] = wmma_bf16(pf, vb, acc[dt]);
        }
        __syncthreads();
    }
#pragma unroll
    for (int r = 0; r < 8; r++) li[r] = 1.0f / li[r];
#pragma unroll
    for (int dt = 0; dt < 8; dt++)
#pragma unroll
        for (int r = 0; r < 8; r++) {
            int m = q0 + r + 8 * hb;
            y[(size_t)m * DIMV + h * HDIM + dt * 16 + n] = f2bf(acc[dt][r] * li[r]);
        }
}

extern "C" void kernel_launch(void* const* d_in, const int* in_sizes, int n_in,
                              void* d_out, int out_size, void* d_ws, size_t ws_size,
                              hipStream_t stream) {
    const float* x       = (const float*)d_in[0];
    const float* ve      = (const float*)d_in[1];
    const float* lambdas = (const float*)d_in[2];
    const float* qkvo    = (const float*)d_in[3];
    const int*   wptr    = (const int*)d_in[4];
    float* out = (float*)d_out;

    char* ws = (char*)d_ws;
    const size_t MB = 1024u * 1024u;
    bfloat* xb   = (bfloat*)(ws + 0 * MB);    // 4096x1024 bf16        (8 MB)
    bfloat* wb   = (bfloat*)(ws + 8 * MB);    // 4x1024x1024 bf16      (8 MB)
    float*  qkvf = (float*)(ws + 16 * MB);    // 4096x3072 f32        (48 MB)
    bfloat* qn   = (bfloat*)(ws + 64 * MB);   // 4096x1024 bf16        (8 MB)
    bfloat* kn   = (bfloat*)(ws + 72 * MB);   // 4096x1024 bf16        (8 MB)
    bfloat* vtb  = (bfloat*)(ws + 80 * MB);   // [8][128][4096] bf16   (8 MB)
    bfloat* yb   = (bfloat*)(ws + 88 * MB);   // 4096x1024 bf16        (8 MB)

    cvt_kernel<<<dim3((T_LEN * DIMV + 255) / 256), 256, 0, stream>>>(x, xb, T_LEN * DIMV);
    cvt_kernel<<<dim3((4 * DIMV * DIMV + 255) / 256), 256, 0, stream>>>(qkvo, wb, 4 * DIMV * DIMV);

    // qkv = x @ Wqkv^T   (M=4096, N=3072, K=1024)
    gemm_nt<<<dim3(T_LEN / 128, 3 * DIMV / 128), 128, 0, stream>>>(xb, wb, qkvf, 3 * DIMV, DIMV);

    normrope<<<dim3(T_LEN, 24), 32, 0, stream>>>(qkvf, ve, lambdas, qn, kn, vtb);

    attn<<<dim3(T_LEN / 16, NHEAD), 32, 0, stream>>>(qn, kn, vtb, yb, wptr);

    // out = y @ Wo^T   (M=4096, N=1024, K=1024)
    gemm_nt<<<dim3(T_LEN / 128, DIMV / 128), 128, 0, stream>>>(yb, wb + 3 * DIMV * DIMV, out, DIMV, DIMV);
}
